// BlockCirculantLinear_53291954209347
// MI455X (gfx1250) — matrix operand order repaired and checked by reading, not compiled, not run on hardware
//
#include <hip/hip_runtime.h>

typedef __attribute__((ext_vector_type(16))) __bf16 v16bf;
typedef __attribute__((ext_vector_type(8)))  float  v8f;

#define M_TOTAL 8192
#define K_TOTAL 4096
#define N_TOTAL 4096
#define BLKB    128                 // circulant block size
#define M_TILE  128
#define N_TILE  128                 // == one o-block
#define K_TILE  64
#define XS_PITCH_F 33               // fallback x-tile pitch (u32 pairs), gcd(33,64)=1
#define XS_PITCH_W 36               // ws x-tile pitch (u32 pairs), 16B multiple
#define PAIRS_N (32 * 256)          // u32 entries per plane (j=0..31, q=0..255)

union Frag { v16bf v; unsigned int u[8]; };

static __device__ __forceinline__ unsigned short f32_to_bf16(float f) {
    union { float f; unsigned int u; } v; v.f = f;
    unsigned int r = v.u + 0x7FFFu + ((v.u >> 16) & 1u);   // round-to-nearest-even
    return (unsigned short)(r >> 16);
}
static __device__ __forceinline__ float bf16_as_f32(unsigned short h) {
    union { unsigned int u; float f; } v; v.u = ((unsigned int)h) << 16;
    return v.f;
}

// ---- CDNA5 async global->LDS copy (ASYNCcnt), portable inline-asm form ----
static __device__ __forceinline__ void async_b128(const void* g, void* l) {
    asm volatile("global_load_async_to_lds_b128 %0, %1, off"
                 :: "v"((__attribute__((address_space(3))) void*)l), "v"(g)
                 : "memory");
}
static __device__ __forceinline__ void wait_async0() {
    asm volatile("s_wait_asynccnt 0x0" ::: "memory");
}

// Build doubled pre-packed circulant pair table for o-block `o`:
//   Pairs[f]          = bf16hi(w[q&127]) | bf16hi(w[(q+1)&127])<<16   (f = j*256+q)
//   Pairs[PAIRS_N+f]  = same for the lo residuals
static __device__ __forceinline__ void build_pairs(unsigned int* Pairs, const float* W,
                                                   int o, int tid) {
    for (int f = tid; f < PAIRS_N; f += 256) {
        int j = f >> 8, q = f & 255;
        const float* wrow = W + (size_t)o * (32 * BLKB) + j * BLKB;
        float w0 = wrow[q & 127];
        float w1 = wrow[(q + 1) & 127];
        unsigned short h0 = f32_to_bf16(w0), h1 = f32_to_bf16(w1);
        unsigned short l0 = f32_to_bf16(w0 - bf16_as_f32(h0));
        unsigned short l1 = f32_to_bf16(w1 - bf16_as_f32(h1));
        Pairs[f]           = (unsigned)h0 | ((unsigned)h1 << 16);
        Pairs[PAIRS_N + f] = (unsigned)l0 | ((unsigned)l1 << 16);
    }
}

// ---------------- precompute: split x*D into bf16 hi/lo planes ----------------
__global__ __launch_bounds__(256) void split_xd(const float* __restrict__ x,
                                                const float* __restrict__ D,
                                                unsigned short* __restrict__ xHi,
                                                unsigned short* __restrict__ xLo) {
    size_t base = ((size_t)blockIdx.x * 256 + threadIdx.x) * 4;
    int k = (int)(base & (K_TOTAL - 1));
    const float4 xv = *(const float4*)(x + base);
    const float4 dv = *(const float4*)(D + k);
    float e0 = xv.x * dv.x, e1 = xv.y * dv.y, e2 = xv.z * dv.z, e3 = xv.w * dv.w;
    unsigned short h0 = f32_to_bf16(e0), h1 = f32_to_bf16(e1);
    unsigned short h2 = f32_to_bf16(e2), h3 = f32_to_bf16(e3);
    unsigned short l0 = f32_to_bf16(e0 - bf16_as_f32(h0));
    unsigned short l1 = f32_to_bf16(e1 - bf16_as_f32(h1));
    unsigned short l2 = f32_to_bf16(e2 - bf16_as_f32(h2));
    unsigned short l3 = f32_to_bf16(e3 - bf16_as_f32(h3));
    uint2 ph; ph.x = (unsigned)h0 | ((unsigned)h1 << 16); ph.y = (unsigned)h2 | ((unsigned)h3 << 16);
    uint2 pl; pl.x = (unsigned)l0 | ((unsigned)l1 << 16); pl.y = (unsigned)l2 | ((unsigned)l3 << 16);
    *(uint2*)(xHi + base) = ph;
    *(uint2*)(xLo + base) = pl;
}

// ---------------- main GEMM, ws path: async-LDS bf16 planes, double buffered ----------------
__global__ __launch_bounds__(256) void gemm_ws(const unsigned short* __restrict__ xHi,
                                               const unsigned short* __restrict__ xLo,
                                               const float* __restrict__ W,
                                               float* __restrict__ out) {
    __shared__ unsigned int Pairs[2 * PAIRS_N];                  // 64 KB
    __shared__ unsigned int Xst[2 * 2 * 128 * XS_PITCH_W];       // 72 KB [buf][plane][row][pair]

    const int tid  = threadIdx.x;
    const int lane = tid & 31;
    const int hh   = lane >> 4;
    const int nl   = lane & 15;
    const int wave = tid >> 5;
    const int wm   = wave & 3;
    const int wn   = wave >> 2;

    const int o  = blockIdx.x;
    const int n0 = o * N_TILE;
    const int m0 = blockIdx.y * M_TILE;

    build_pairs(Pairs, W, o, tid);

    int vbq[4];
    #pragma unroll
    for (int sn = 0; sn < 4; ++sn)
        vbq[sn] = 128 - (wn * 64 + sn * 16 + nl) + 16 * hh;      // >= 1

    int mrowA[2];
    #pragma unroll
    for (int sm = 0; sm < 2; ++sm) mrowA[sm] = wm * 32 + sm * 16 + nl;

    v8f acc[2][4];
    #pragma unroll
    for (int a = 0; a < 2; ++a)
        #pragma unroll
        for (int b = 0; b < 4; ++b)
            #pragma unroll
            for (int e = 0; e < 8; ++e) acc[a][b][e] = 0.0f;

    const unsigned short* planes[2] = { xHi, xLo };
    auto copy_tile = [&](int kt, int nbuf) {
        #pragma unroll
        for (int pl = 0; pl < 2; ++pl)
            #pragma unroll
            for (int it = 0; it < 4; ++it) {
                int L = tid + it * 256;
                int row = L >> 3, ch = L & 7;                    // 8 * b128 per 64-wide row
                const unsigned short* g = planes[pl] + (size_t)(m0 + row) * K_TOTAL + kt + ch * 8;
                unsigned int* l = &Xst[((nbuf * 2 + pl) * 128 + row) * XS_PITCH_W + ch * 4];
                async_b128(g, l);
            }
    };

    copy_tile(0, 0);                                             // prefetch tile 0

    int buf = 0;
    for (int k0 = 0; k0 < K_TOTAL; k0 += K_TILE, buf ^= 1) {
        wait_async0();
        __syncthreads();                                         // tile(buf) ready; buf^1 free
        if (k0 + K_TILE < K_TOTAL) copy_tile(k0 + K_TILE, buf ^ 1);

        const int jb = (k0 >> 7) * 256 + (k0 & 127);             // Pairs scalar base (u32 idx)
        const int xb = buf * 2 * 128 * XS_PITCH_W;               // x-tile scalar base (u32 idx)

        #pragma unroll
        for (int kk = 0; kk < 2; ++kk) {
            // A fragments: pair idx p = e + 8*half
            Frag aH[2], aL[2];
            #pragma unroll
            for (int sm = 0; sm < 2; ++sm) {
                const unsigned int* pA = &Xst[xb + mrowA[sm] * XS_PITCH_W + kk * 16];
                #pragma unroll
                for (int e = 0; e < 8; ++e) {
                    int p = e + 8 * hh;
                    aH[sm].u[e] = pA[p];
                    aL[sm].u[e] = pA[128 * XS_PITCH_W + p];      // lo plane, +18432B imm
                }
            }
            // B fragments: one b32 each from doubled pair table, imm offsets 8e / 32768+8e
            Frag bH[4], bL[4];
            #pragma unroll
            for (int sn = 0; sn < 4; ++sn) {
                const unsigned int* pB = &Pairs[jb + kk * 32 + vbq[sn]];
                #pragma unroll
                for (int e = 0; e < 8; ++e) {
                    bH[sn].u[e] = pB[2 * e];
                    bL[sn].u[e] = pB[PAIRS_N + 2 * e];
                }
            }
            #pragma unroll
            for (int sm = 0; sm < 2; ++sm)
                #pragma unroll
                for (int sn = 0; sn < 4; ++sn) {
                    acc[sm][sn] = __builtin_amdgcn_wmma_f32_16x16x32_bf16(
                        false, aH[sm].v, false, bH[sn].v, (short)0, acc[sm][sn], false, false);
                    acc[sm][sn] = __builtin_amdgcn_wmma_f32_16x16x32_bf16(
                        false, aH[sm].v, false, bL[sn].v, (short)0, acc[sm][sn], false, false);
                    acc[sm][sn] = __builtin_amdgcn_wmma_f32_16x16x32_bf16(
                        false, aL[sm].v, false, bH[sn].v, (short)0, acc[sm][sn], false, false);
                }
        }
    }

    #pragma unroll
    for (int sm = 0; sm < 2; ++sm)
        #pragma unroll
        for (int sn = 0; sn < 4; ++sn)
            #pragma unroll
            for (int r = 0; r < 8; ++r) {
                int m = m0 + wm * 32 + sm * 16 + r + 8 * hh;
                int n = n0 + wn * 64 + sn * 16 + nl;
                out[(size_t)m * N_TOTAL + n] = acc[sm][sn][r];
            }
}

// ---------------- fallback: self-contained (no workspace), in-kernel split ----------------
__global__ __launch_bounds__(256) void gemm_fb(const float* __restrict__ x,
                                               const float* __restrict__ W,
                                               const float* __restrict__ D,
                                               float* __restrict__ out) {
    __shared__ unsigned int Pairs[2 * PAIRS_N];                  // 64 KB
    __shared__ unsigned int Xsp[2 * 128 * XS_PITCH_F];           // 33 KB [plane][row][pair]

    const int tid  = threadIdx.x;
    const int lane = tid & 31;
    const int hh   = lane >> 4;
    const int nl   = lane & 15;
    const int wave = tid >> 5;
    const int wm   = wave & 3;
    const int wn   = wave >> 2;

    const int o  = blockIdx.x;
    const int n0 = o * N_TILE;
    const int m0 = blockIdx.y * M_TILE;

    build_pairs(Pairs, W, o, tid);

    int vbq[4];
    #pragma unroll
    for (int sn = 0; sn < 4; ++sn)
        vbq[sn] = 128 - (wn * 64 + sn * 16 + nl) + 16 * hh;

    int mrowA[2];
    #pragma unroll
    for (int sm = 0; sm < 2; ++sm) mrowA[sm] = wm * 32 + sm * 16 + nl;

    v8f acc[2][4];
    #pragma unroll
    for (int a = 0; a < 2; ++a)
        #pragma unroll
        for (int b = 0; b < 4; ++b)
            #pragma unroll
            for (int e = 0; e < 8; ++e) acc[a][b][e] = 0.0f;

    const int row = tid >> 4;
    const int c4  = tid & 15;

    for (int k0 = 0; k0 < K_TOTAL; k0 += K_TILE) {
        const float4 dv = *(const float4*)(D + k0 + c4 * 4);     // hoisted per tile
        #pragma unroll
        for (int it = 0; it < 8; ++it) {
            int r = row + it * 16;
            const float4 xv = *(const float4*)(x + (size_t)(m0 + r) * K_TOTAL + k0 + c4 * 4);
            float e0 = xv.x * dv.x, e1 = xv.y * dv.y, e2 = xv.z * dv.z, e3 = xv.w * dv.w;
            unsigned short h0 = f32_to_bf16(e0), h1 = f32_to_bf16(e1);
            unsigned short h2 = f32_to_bf16(e2), h3 = f32_to_bf16(e3);
            unsigned short l0 = f32_to_bf16(e0 - bf16_as_f32(h0));
            unsigned short l1 = f32_to_bf16(e1 - bf16_as_f32(h1));
            unsigned short l2 = f32_to_bf16(e2 - bf16_as_f32(h2));
            unsigned short l3 = f32_to_bf16(e3 - bf16_as_f32(h3));
            unsigned int* pH = &Xsp[r * XS_PITCH_F + c4 * 2];
            pH[0] = (unsigned)h0 | ((unsigned)h1 << 16);
            pH[1] = (unsigned)h2 | ((unsigned)h3 << 16);
            pH[128 * XS_PITCH_F + 0] = (unsigned)l0 | ((unsigned)l1 << 16);
            pH[128 * XS_PITCH_F + 1] = (unsigned)l2 | ((unsigned)l3 << 16);
        }
        __syncthreads();

        const int jb = (k0 >> 7) * 256 + (k0 & 127);

        #pragma unroll
        for (int kk = 0; kk < 2; ++kk) {
            Frag aH[2], aL[2];
            #pragma unroll
            for (int sm = 0; sm < 2; ++sm) {
                const unsigned int* pA = &Xsp[mrowA[sm] * XS_PITCH_F + kk * 16];
                #pragma unroll
                for (int e = 0; e < 8; ++e) {
                    int p = e + 8 * hh;
                    aH[sm].u[e] = pA[p];
                    aL[sm].u[e] = pA[128 * XS_PITCH_F + p];
                }
            }
            Frag bH[4], bL[4];
            #pragma unroll
            for (int sn = 0; sn < 4; ++sn) {
                const unsigned int* pB = &Pairs[jb + kk * 32 + vbq[sn]];
                #pragma unroll
                for (int e = 0; e < 8; ++e) {
                    bH[sn].u[e] = pB[2 * e];
                    bL[sn].u[e] = pB[PAIRS_N + 2 * e];
                }
            }
            #pragma unroll
            for (int sm = 0; sm < 2; ++sm)
                #pragma unroll
                for (int sn = 0; sn < 4; ++sn) {
                    acc[sm][sn] = __builtin_amdgcn_wmma_f32_16x16x32_bf16(
                        false, aH[sm].v, false, bH[sn].v, (short)0, acc[sm][sn], false, false);
                    acc[sm][sn] = __builtin_amdgcn_wmma_f32_16x16x32_bf16(
                        false, aH[sm].v, false, bL[sn].v, (short)0, acc[sm][sn], false, false);
                    acc[sm][sn] = __builtin_amdgcn_wmma_f32_16x16x32_bf16(
                        false, aL[sm].v, false, bH[sn].v, (short)0, acc[sm][sn], false, false);
                }
        }
        __syncthreads();
    }

    #pragma unroll
    for (int sm = 0; sm < 2; ++sm)
        #pragma unroll
        for (int sn = 0; sn < 4; ++sn)
            #pragma unroll
            for (int r = 0; r < 8; ++r) {
                int m = m0 + wm * 32 + sm * 16 + r + 8 * hh;
                int n = n0 + wn * 64 + sn * 16 + nl;
                out[(size_t)m * N_TOTAL + n] = acc[sm][sn][r];
            }
}

extern "C" void kernel_launch(void* const* d_in, const int* in_sizes, int n_in,
                              void* d_out, int out_size, void* d_ws, size_t ws_size,
                              hipStream_t stream) {
    (void)in_sizes; (void)n_in; (void)out_size;
    const float* x = (const float*)d_in[0];
    const float* W = (const float*)d_in[1];
    const float* D = (const float*)d_in[2];
    float* out     = (float*)d_out;

    const size_t planeElems = (size_t)M_TOTAL * K_TOTAL;
    const size_t need = planeElems * 2 * sizeof(unsigned short);   // 134,217,728 B
    dim3 grid(N_TOTAL / N_TILE, M_TOTAL / M_TILE);                 // 32 x 64

    if (ws_size >= need) {
        unsigned short* xHi = (unsigned short*)d_ws;
        unsigned short* xLo = xHi + planeElems;
        split_xd<<<(unsigned)(planeElems / 4 / 256), 256, 0, stream>>>(x, D, xHi, xLo);
        gemm_ws<<<grid, 256, 0, stream>>>(xHi, xLo, W, out);
    } else {
        gemm_fb<<<grid, 256, 0, stream>>>(x, W, D, out);
    }
}
